// CAM_Module_37306085933412
// MI455X (gfx1250) — compile-verified, matrix operand order checked by reading, not run on hardware
//
#include <hip/hip_runtime.h>
#include <hip/hip_bf16.h>

typedef __attribute__((ext_vector_type(16))) __bf16 v16bf;
typedef __attribute__((ext_vector_type(8)))  __bf16 v8bf;
typedef __attribute__((ext_vector_type(4)))  __bf16 v4bf;
typedef __attribute__((ext_vector_type(8)))  float  v8f;
typedef __attribute__((ext_vector_type(4)))  float  v4f;

#define C_DIM 512
#define N_DIM 4096
#define LDSLD 40   // padded LDS row stride in bf16 elements (80B: conflict-free frag loads)

// Use GLOBAL_LOAD_ASYNC_TO_LDS_B128 (ASYNCcnt path) for the bf16 attention tile.
#define USE_GFX1250_ASYNC 1

// Load one 16x32 bf16 WMMA operand fragment from an LDS tile stored row-major.
// Per ISA 7.12.2: lane m (0-15) holds K [0..7],[16..23]; lane m+16 holds K [8..15],[24..31].
__device__ __forceinline__ v16bf load_frag(const __bf16* row0, int m, int hi) {
  const v8bf* p0 = (const v8bf*)(row0 + m * LDSLD + hi * 8);
  const v8bf* p1 = (const v8bf*)(row0 + m * LDSLD + 16 + hi * 8);
  v8bf a = *p0;
  v8bf b = *p1;
  return __builtin_shufflevector(a, b, 0,1,2,3,4,5,6,7,8,9,10,11,12,13,14,15);
}

// 8 WMMAs of one wave's 32x64 sub-tile against a staged 128x32 K-slice.
__device__ __forceinline__ void tile_compute(const __bf16* As, const __bf16* Bs,
                                             int wm, int wn, int m, int hi,
                                             v8f acc[2][4]) {
  v16bf af[2], bfr[4];
#pragma unroll
  for (int a = 0; a < 2; ++a) af[a] = load_frag(As + (wm + a * 16) * LDSLD, m, hi);
#pragma unroll
  for (int c = 0; c < 4; ++c) bfr[c] = load_frag(Bs + (wn + c * 16) * LDSLD, m, hi);
#pragma unroll
  for (int a = 0; a < 2; ++a)
#pragma unroll
    for (int c = 0; c < 4; ++c)
      acc[a][c] = __builtin_amdgcn_wmma_f32_16x16x32_bf16(
          false, af[a], false, bfr[c], (short)0, acc[a][c], false, false);
}

// ---------------------------------------------------------------------------
// Kernel 1: E[b] = Q[b] * Q[b]^T.  Both WMMA operands are rows of Q, so no
// transpose anywhere. Double-buffered LDS + register prefetch: global loads of
// step k+1 are issued before the WMMAs of step k; one barrier per K-step.
// ---------------------------------------------------------------------------
__device__ __forceinline__ void g1_load(const float* xb, int i0, int j0, int kk,
                                        int lr, int lc, v4f pa[4], v4f pb[4]) {
#pragma unroll
  for (int rr = 0; rr < 4; ++rr) {
    const int row = lr + rr * 32;
    pa[rr] = *(const v4f*)(xb + (size_t)(i0 + row) * N_DIM + kk + lc);
    pb[rr] = *(const v4f*)(xb + (size_t)(j0 + row) * N_DIM + kk + lc);
  }
}

__device__ __forceinline__ void g1_store(__bf16* As, __bf16* Bs, int lr, int lc,
                                         const v4f pa[4], const v4f pb[4]) {
#pragma unroll
  for (int rr = 0; rr < 4; ++rr) {
    const int row = lr + rr * 32;
    v4bf ca, cb;
#pragma unroll
    for (int j = 0; j < 4; ++j) { ca[j] = (__bf16)pa[rr][j]; cb[j] = (__bf16)pb[rr][j]; }
    *(v4bf*)(As + row * LDSLD + lc) = ca;
    *(v4bf*)(Bs + row * LDSLD + lc) = cb;
  }
}

__global__ void __launch_bounds__(256)
cam_gram_kernel(const float* __restrict__ x, float* __restrict__ energy) {
  __shared__ __align__(16) __bf16 As[2][128 * LDSLD];
  __shared__ __align__(16) __bf16 Bs[2][128 * LDSLD];

  const int b    = blockIdx.z;
  const int i0   = blockIdx.y * 128;
  const int j0   = blockIdx.x * 128;
  const int tid  = threadIdx.x;
  const int lane = tid & 31;
  const int wave = tid >> 5;
  const int m    = lane & 15;
  const int hi   = lane >> 4;
  const int wm   = (wave >> 1) * 32;
  const int wn   = (wave & 1) * 64;
  const int lr   = tid >> 3;
  const int lc   = (tid & 7) * 4;

  const float* xb = x + (size_t)b * C_DIM * N_DIM;

  v8f zero = {};
  v8f acc[2][4];
#pragma unroll
  for (int a = 0; a < 2; ++a)
#pragma unroll
    for (int c = 0; c < 4; ++c) acc[a][c] = zero;

  v4f pa[4], pb[4];
  g1_load(xb, i0, j0, 0, lr, lc, pa, pb);
  g1_store(As[0], Bs[0], lr, lc, pa, pb);
  __syncthreads();

  int buf = 0;
  for (int kk = 32; kk < N_DIM; kk += 32) {
    g1_load(xb, i0, j0, kk, lr, lc, pa, pb);           // prefetch (in flight)
    tile_compute(As[buf], Bs[buf], wm, wn, m, hi, acc); // overlaps VMEM
    g1_store(As[buf ^ 1], Bs[buf ^ 1], lr, lc, pa, pb); // waits loadcnt here
    __syncthreads();
    buf ^= 1;
  }
  tile_compute(As[buf], Bs[buf], wm, wn, m, hi, acc);

  float* Eb = energy + (size_t)b * C_DIM * C_DIM;
#pragma unroll
  for (int a = 0; a < 2; ++a)
#pragma unroll
    for (int c = 0; c < 4; ++c)
#pragma unroll
      for (int v = 0; v < 8; ++v) {
        const int row = i0 + wm + a * 16 + hi * 8 + v;
        const int col = j0 + wn + c * 16 + m;
        Eb[(size_t)row * C_DIM + col] = acc[a][c][v];
      }
}

// ---------------------------------------------------------------------------
// Kernel 2: row softmax. softmax(rowmax - E) == exp(rowmin - E)/sum, so only a
// row-min is needed. One wave per row, 16 elems/lane in registers, wave32
// xor-reductions via ds_swizzle (group-of-32 mode), b128 row I/O.
// ---------------------------------------------------------------------------
#define SWZF(v, mask) __uint_as_float(__builtin_amdgcn_ds_swizzle(__float_as_uint(v), ((mask) << 10) | 0x1f))

__device__ __forceinline__ float wave_min_f32(float v) {
  v = fminf(v, SWZF(v, 16));
  v = fminf(v, SWZF(v, 8));
  v = fminf(v, SWZF(v, 4));
  v = fminf(v, SWZF(v, 2));
  v = fminf(v, SWZF(v, 1));
  return v;
}
__device__ __forceinline__ float wave_sum_f32(float v) {
  v += SWZF(v, 16);
  v += SWZF(v, 8);
  v += SWZF(v, 4);
  v += SWZF(v, 2);
  v += SWZF(v, 1);
  return v;
}

__global__ void __launch_bounds__(256)
cam_softmax_kernel(const float* __restrict__ energy, __bf16* __restrict__ att) {
  const int wave = threadIdx.x >> 5;
  const int lane = threadIdx.x & 31;
  const int row  = blockIdx.x * 8 + wave;          // 0 .. 16*512-1
  const v4f* Er4 = (const v4f*)(energy + (size_t)row * C_DIM);

  float v[16];
  float mn = 3.402823466e+38f;
#pragma unroll
  for (int t = 0; t < 4; ++t) {
    v4f c = Er4[lane + 32 * t];
#pragma unroll
    for (int j = 0; j < 4; ++j) { v[4 * t + j] = c[j]; mn = fminf(mn, c[j]); }
  }
  mn = wave_min_f32(mn);

  float s = 0.0f;
#pragma unroll
  for (int t = 0; t < 16; ++t) {
    v[t] = __expf(mn - v[t]);
    s += v[t];
  }
  s = wave_sum_f32(s);
  const float inv = 1.0f / s;   // one element equals exp(0)=1, so s >= 1

  v4bf* Ar4 = (v4bf*)(att + (size_t)row * C_DIM);
#pragma unroll
  for (int t = 0; t < 4; ++t) {
    v4bf c;
#pragma unroll
    for (int j = 0; j < 4; ++j) c[j] = (__bf16)(v[4 * t + j] * inv);
    Ar4[lane + 32 * t] = c;
  }
}

// ---------------------------------------------------------------------------
// Kernel 3: out = gamma * (A @ Q) + x.  M=512, N=4096, K=512 per batch.
// Attention tile (pure bf16 copy) goes global->LDS via GLOBAL_LOAD_ASYNC_TO_LDS
// (ASYNCcnt); Q tile is f32->bf16 converted + transposed through registers.
// Same double-buffer pipeline as kernel 1.
// ---------------------------------------------------------------------------
__device__ __forceinline__ void g2_loadA_async(const __bf16* abase /* ab + kk */,
                                               __bf16* Adst, int a_r, int a_ch) {
#pragma unroll
  for (int rr = 0; rr < 2; ++rr) {
    const int row = a_r + rr * 64;
    const unsigned voff = (unsigned)((row * C_DIM + a_ch) * 2);               // bytes
    const unsigned loff = (unsigned)(size_t)(Adst + row * LDSLD + a_ch);      // LDS byte addr
    asm volatile("global_load_async_to_lds_b128 %0, %1, %2"
                 :
                 : "v"(loff), "v"(voff), "s"(abase)
                 : "memory");
  }
}

__device__ __forceinline__ void g2_loadB(const float* xb, int n0, int kk,
                                         int b_c, int b_d, v4f pf[4]) {
#pragma unroll
  for (int it = 0; it < 4; ++it) {
    const int d = b_d + it * 8;
    pf[it] = *(const v4f*)(xb + (size_t)(kk + d) * N_DIM + n0 + b_c);
  }
}

__device__ __forceinline__ void g2_storeB(__bf16* Bdst, int b_c, int b_d, const v4f pf[4]) {
#pragma unroll
  for (int it = 0; it < 4; ++it) {
    const int d = b_d + it * 8;
#pragma unroll
    for (int j = 0; j < 4; ++j) Bdst[(b_c + j) * LDSLD + d] = (__bf16)pf[it][j];
  }
}

__global__ void __launch_bounds__(256)
cam_av_kernel(const float* __restrict__ x, const __bf16* __restrict__ att,
              const float* __restrict__ gamma, float* __restrict__ out) {
  __shared__ __align__(16) __bf16 As[2][128 * LDSLD];  // attention tile, [M][K]
  __shared__ __align__(16) __bf16 Bs[2][128 * LDSLD];  // Q tile transposed, [N][K]

  const int b    = blockIdx.z;
  const int i0   = blockIdx.y * 128;   // M (channel out)
  const int n0   = blockIdx.x * 128;   // N (pixel)
  const int tid  = threadIdx.x;
  const int lane = tid & 31;
  const int wave = tid >> 5;
  const int m    = lane & 15;
  const int hi   = lane >> 4;
  const int wm   = (wave >> 1) * 32;
  const int wn   = (wave & 1) * 64;

  const float*  xb = x   + (size_t)b * C_DIM * N_DIM;
  const __bf16* ab = att + (size_t)b * C_DIM * C_DIM + (size_t)i0 * C_DIM;

  v8f zero = {};
  v8f acc[2][4];
#pragma unroll
  for (int a = 0; a < 2; ++a)
#pragma unroll
    for (int c = 0; c < 4; ++c) acc[a][c] = zero;

  const int a_ch = (tid & 3) * 8;   // K offset for A loads (8 bf16 = 16B)
  const int a_r  = tid >> 2;        // 0..63
  const int b_c  = (tid & 31) * 4;  // N offset for B loads (4 floats)
  const int b_d  = tid >> 5;        // 0..7  (K row)

  v4f pf[4];
#if USE_GFX1250_ASYNC
  g2_loadA_async(ab, As[0], a_r, a_ch);
#else
#pragma unroll
  for (int rr = 0; rr < 2; ++rr) {
    const int row = a_r + rr * 64;
    *(v8bf*)(As[0] + row * LDSLD + a_ch) = *(const v8bf*)(ab + (size_t)row * C_DIM + a_ch);
  }
#endif
  g2_loadB(xb, n0, 0, b_c, b_d, pf);
  g2_storeB(Bs[0], b_c, b_d, pf);
#if USE_GFX1250_ASYNC
  asm volatile("s_wait_asynccnt 0x0" ::: "memory");
#endif
  __syncthreads();

  int buf = 0;
  for (int kk = 32; kk < C_DIM; kk += 32) {
#if USE_GFX1250_ASYNC
    g2_loadA_async(ab + kk, As[buf ^ 1], a_r, a_ch);        // async -> LDS[next]
#else
#pragma unroll
    for (int rr = 0; rr < 2; ++rr) {
      const int row = a_r + rr * 64;
      *(v8bf*)(As[buf ^ 1] + row * LDSLD + a_ch) =
          *(const v8bf*)(ab + (size_t)row * C_DIM + kk + a_ch);
    }
#endif
    g2_loadB(xb, n0, kk, b_c, b_d, pf);                     // prefetch (in flight)
    tile_compute(As[buf], Bs[buf], wm, wn, m, hi, acc);     // overlaps VMEM/async
    g2_storeB(Bs[buf ^ 1], b_c, b_d, pf);
#if USE_GFX1250_ASYNC
    asm volatile("s_wait_asynccnt 0x0" ::: "memory");
#endif
    __syncthreads();
    buf ^= 1;
  }
  tile_compute(As[buf], Bs[buf], wm, wn, m, hi, acc);

  const float g = gamma[0];
#pragma unroll
  for (int a = 0; a < 2; ++a)
#pragma unroll
    for (int c = 0; c < 4; ++c)
#pragma unroll
      for (int v = 0; v < 8; ++v) {
        const int row = i0 + wm + a * 16 + hi * 8 + v;
        const int col = n0 + wn + c * 16 + m;
        const size_t idx = (size_t)b * C_DIM * N_DIM + (size_t)row * N_DIM + col;
        out[idx] = g * acc[a][c][v] + x[idx];
      }
}

// ---------------------------------------------------------------------------
extern "C" void kernel_launch(void* const* d_in, const int* in_sizes, int n_in,
                              void* d_out, int out_size, void* d_ws, size_t ws_size,
                              hipStream_t stream) {
  const float* x     = (const float*)d_in[0];
  const float* gamma = (const float*)d_in[1];
  float*       out   = (float*)d_out;

  // workspace: [0, 16MB) energy f32 (16x512x512), [16MB, 24MB) attention bf16
  float*  energy = (float*)d_ws;
  __bf16* att    = (__bf16*)((char*)d_ws + (size_t)16 * C_DIM * C_DIM * sizeof(float));

  dim3 blk(256);
  cam_gram_kernel<<<dim3(4, 4, 16), blk, 0, stream>>>(x, energy);
  cam_softmax_kernel<<<dim3(1024), blk, 0, stream>>>(energy, att);
  cam_av_kernel<<<dim3(32, 4, 16), blk, 0, stream>>>(x, att, gamma, out);
}
